// MoEUTTimeSeriesDecoder_38285338477274
// MI455X (gfx1250) — matrix operand order repaired and checked by reading, not compiled, not run on hardware
//
#include <hip/hip_runtime.h>

typedef __bf16 bf16;
typedef __attribute__((ext_vector_type(16))) __bf16 v16bf;
typedef __attribute__((ext_vector_type(8)))  float  v8f;

#define NLAYER 3
#define DMODEL 1024
#define NHEAD  16
#define HDIM   64
#define NEXP   8
#define EHID   256
#define BATCH  2
#define SEQ    1024
#define DIN    32
#define NOUTD  8
#define NTOK   (BATCH*SEQ)

// ---------------------------------------------------------------------------
// WMMA fragment loaders (CDNA5 §7.12.2 layouts, wave32)
// A (16x32 bf16): lane = 16*half + l16, row m=l16; K = 8*half+{0..7} and +16
// B via Bt[N][K]: lane col n=l16; K = 8*half+{0..7} and +16
// ---------------------------------------------------------------------------
union Frag16 { uint4 q[2]; v16bf v; };

__device__ __forceinline__ v16bf ldA(const bf16* A, int lda, int m, int k0, int half) {
  Frag16 f;
  const bf16* p = A + (size_t)m * lda + k0 + 8 * half;
  f.q[0] = *(const uint4*)p;
  f.q[1] = *(const uint4*)(p + 16);
  return f.v;
}
__device__ __forceinline__ v16bf ldB(const bf16* Bt, int ldb, int n, int k0, int half) {
  Frag16 f;
  const bf16* p = Bt + (size_t)n * ldb + k0 + 8 * half;
  f.q[0] = *(const uint4*)p;
  f.q[1] = *(const uint4*)(p + 16);
  return f.v;
}
#define WMMA_BF16(a, b, c) \
  __builtin_amdgcn_wmma_f32_16x16x32_bf16(false, (a), false, (b), (short)0, (c), false, false)

// ---------------------------------------------------------------------------
// Generic bf16 WMMA GEMM: C[M,N](f32 or bf16) = A[M,K] @ Bt[N,K]^T (+epilogue)
// block = 256 threads = 8 waves (2x4) -> 64x128 block tile
// wave = 32x32 tile = 2x2 WMMA tiles (each frag feeds 2 WMMAs, 4 indep. chains)
// ---------------------------------------------------------------------------
__global__ __launch_bounds__(256)
void k_gemm(const bf16* __restrict__ A, const bf16* __restrict__ Bt,
            int M, int N, int K,
            const float* __restrict__ bias,
            const float* __restrict__ residual,
            const float* __restrict__ rowscale, int rs_stride,
            int relu,
            float* __restrict__ outF, bf16* __restrict__ outB)
{
  const int tid  = threadIdx.x;
  const int wave = tid >> 5, lane = tid & 31;
  const int half = lane >> 4, l16 = lane & 15;
  const int wm = wave >> 2, wn = wave & 3;
  const int tm = blockIdx.y * 64  + wm * 32;
  const int tn = blockIdx.x * 128 + wn * 32;

  v8f acc00 = {0.f,0.f,0.f,0.f,0.f,0.f,0.f,0.f};
  v8f acc01 = acc00, acc10 = acc00, acc11 = acc00;

  const int am0 = tm + l16,      am1 = tm + 16 + l16;
  const int bn0 = tn + l16,      bn1 = tn + 16 + l16;

  for (int k0 = 0; k0 < K; k0 += 32) {
    v16bf a0 = ldA(A,  K, am0, k0, half);
    v16bf a1 = ldA(A,  K, am1, k0, half);
    v16bf b0 = ldB(Bt, K, bn0, k0, half);
    v16bf b1 = ldB(Bt, K, bn1, k0, half);
    if (k0 + 32 < K) {  // WGP-scope prefetch of next K-slice (locality 3)
      __builtin_prefetch(A  + (size_t)am0 * K + k0 + 32, 0, 3);
      __builtin_prefetch(A  + (size_t)am1 * K + k0 + 32, 0, 3);
      __builtin_prefetch(Bt + (size_t)bn0 * K + k0 + 32, 0, 3);
      __builtin_prefetch(Bt + (size_t)bn1 * K + k0 + 32, 0, 3);
    }
    acc00 = WMMA_BF16(a0, b0, acc00);
    acc01 = WMMA_BF16(a0, b1, acc01);
    acc10 = WMMA_BF16(a1, b0, acc10);
    acc11 = WMMA_BF16(a1, b1, acc11);
  }

  const float bv0 = bias ? bias[bn0] : 0.f;
  const float bv1 = bias ? bias[bn1] : 0.f;
#pragma unroll
  for (int mt = 0; mt < 2; ++mt) {
    v8f accA = mt ? acc10 : acc00;
    v8f accB = mt ? acc11 : acc01;
#pragma unroll
    for (int r = 0; r < 8; ++r) {
      const int m = tm + mt * 16 + r + 8 * half;
      const float rsv = rowscale ? rowscale[(size_t)m * rs_stride] : 1.f;
#pragma unroll
      for (int nt = 0; nt < 2; ++nt) {
        const int n = (nt ? bn1 : bn0);
        float v = (nt ? accB[r] : accA[r]) + (nt ? bv1 : bv0);
        if (relu)     v = fmaxf(v, 0.f);
        if (rowscale) v *= rsv;
        if (residual) v += residual[(size_t)m * N + n];
        if (outF) outF[(size_t)m * N + n] = v;
        else      outB[(size_t)m * N + n] = (bf16)v;
      }
    }
  }
}

// ---------------------------------------------------------------------------
// Attention: one workgroup per (b, h, 32-row query block). Full-row softmax
// held in LDS (32x1024 f32 + 32x1024 bf16 + Q tile = 196 KB < 320 KB WGP LDS)
// ---------------------------------------------------------------------------
__global__ __launch_bounds__(256)
void k_attn(const bf16* __restrict__ Q, const bf16* __restrict__ Kb,
            const bf16* __restrict__ Vt, bf16* __restrict__ O)
{
  extern __shared__ char smem[];
  bf16*  sQ = (bf16*)smem;                                   // 32*64 bf16
  float* sS = (float*)(smem + 32*64*2);                      // 32*1024 f32
  bf16*  sP = (bf16*)(smem + 32*64*2 + 32*1024*4);           // 32*1024 bf16

  const int tid  = threadIdx.x;
  const int wave = tid >> 5, lane = tid & 31;
  const int half = lane >> 4, l16 = lane & 15;

  int id = blockIdx.x;
  const int qb = id & 31; id >>= 5;
  const int h  = id & 15; id >>= 4;
  const int b  = id;

  const bf16* Qb = Q  + (((size_t)b * NHEAD + h) * SEQ + qb * 32) * HDIM;
  const bf16* Kh = Kb + ((size_t)b * NHEAD + h) * SEQ * HDIM;
  const bf16* Vh = Vt + ((size_t)b * NHEAD + h) * HDIM * SEQ;

  // stage 32x64 bf16 Q tile (4096 B, 256 x uint4)
  ((uint4*)sQ)[tid] = ((const uint4*)Qb)[tid];
  __syncthreads();

  // ---- phase 1: S = (Q K^T) * scale -------------------------------------
  const int rt = wave & 1;
  // Q fragments are invariant across the 16 column tiles: load once.
  const v16bf qa0 = ldA(sQ, HDIM, rt * 16 + l16, 0,  half);
  const v16bf qa1 = ldA(sQ, HDIM, rt * 16 + l16, 32, half);
  for (int ct = wave >> 1; ct < 64; ct += 4) {
    v8f acc = {0.f,0.f,0.f,0.f,0.f,0.f,0.f,0.f};
    v16bf kb0 = ldB(Kh, HDIM, ct * 16 + l16, 0,  half);
    v16bf kb1 = ldB(Kh, HDIM, ct * 16 + l16, 32, half);
    acc = WMMA_BF16(qa0, kb0, acc);
    acc = WMMA_BF16(qa1, kb1, acc);
#pragma unroll
    for (int r = 0; r < 8; ++r)
      sS[(rt * 16 + r + 8 * half) * SEQ + ct * 16 + l16] = acc[r] * 0.125f;
  }
  __syncthreads();

  // ---- phase 2: row softmax (wave-parallel, 4 rows/wave) ----------------
  for (int rr = 0; rr < 4; ++rr) {
    const int row = wave * 4 + rr;
    float mx = -3.4e38f;
    for (int j = lane; j < SEQ; j += 32) mx = fmaxf(mx, sS[row * SEQ + j]);
    for (int o = 16; o > 0; o >>= 1) mx = fmaxf(mx, __shfl_xor(mx, o, 32));
    float sum = 0.f;
    for (int j = lane; j < SEQ; j += 32) {
      float e = __expf(sS[row * SEQ + j] - mx);
      sS[row * SEQ + j] = e;
      sum += e;
    }
    for (int o = 16; o > 0; o >>= 1) sum += __shfl_xor(sum, o, 32);
    const float inv = 1.f / sum;
    for (int j = lane; j < SEQ; j += 32)
      sP[row * SEQ + j] = (bf16)(sS[row * SEQ + j] * inv);
  }
  __syncthreads();

  // ---- phase 3: O = P @ V (V pre-transposed [d][s]) ---------------------
  const int cv = wave >> 1;  // 4 column tiles of 16 over HDIM=64
  {
    v8f acc = {0.f,0.f,0.f,0.f,0.f,0.f,0.f,0.f};
    for (int k0 = 0; k0 < SEQ; k0 += 32) {
      v16bf a  = ldA(sP, SEQ, rt * 16 + l16, k0, half);
      v16bf bb = ldB(Vh, SEQ, cv * 16 + l16, k0, half);
      acc = WMMA_BF16(a, bb, acc);
    }
#pragma unroll
    for (int r = 0; r < 8; ++r) {
      const int row = qb * 32 + rt * 16 + r + 8 * half;
      const int col = h * HDIM + cv * 16 + l16;
      O[((size_t)b * SEQ + row) * DMODEL + col] = (bf16)acc[r];
    }
  }
}

// ---------------------------------------------------------------------------
// LayerNorm -> bf16 (one wave per token, 8 tokens per 256-thread block)
// ---------------------------------------------------------------------------
__global__ __launch_bounds__(256)
void k_ln(const float* __restrict__ X, const float* __restrict__ g,
          const float* __restrict__ bparm, bf16* __restrict__ out)
{
  const int wave = threadIdx.x >> 5, lane = threadIdx.x & 31;
  const int token = blockIdx.x * 8 + wave;
  const float* xr = X + (size_t)token * DMODEL;
  float xs[32]; float s = 0.f;
#pragma unroll
  for (int j = 0; j < 32; ++j) { xs[j] = xr[lane + j * 32]; s += xs[j]; }
  for (int o = 16; o > 0; o >>= 1) s += __shfl_xor(s, o, 32);
  const float mean = s * (1.f / DMODEL);
  float v = 0.f;
#pragma unroll
  for (int j = 0; j < 32; ++j) { float d = xs[j] - mean; v += d * d; }
  for (int o = 16; o > 0; o >>= 1) v += __shfl_xor(v, o, 32);
  const float rs = rsqrtf(v * (1.f / DMODEL) + 1e-5f);
#pragma unroll
  for (int j = 0; j < 32; ++j) {
    const int i = lane + j * 32;
    out[(size_t)token * DMODEL + i] = (bf16)((xs[j] - mean) * rs * g[i] + bparm[i]);
  }
}

// ---------------------------------------------------------------------------
// f32 -> bf16 converters (plain and transposed [batch][R][C] -> [batch][C][R])
// ---------------------------------------------------------------------------
__global__ void k_cvt(const float* __restrict__ in, bf16* __restrict__ out, int n)
{
  const int i = blockIdx.x * blockDim.x + threadIdx.x;
  if (i < n) out[i] = (bf16)in[i];
}
__global__ void k_cvt_tr(const float* __restrict__ in, bf16* __restrict__ out,
                         int n, int R, int C)
{
  const int i = blockIdx.x * blockDim.x + threadIdx.x;
  if (i >= n) return;
  const int c  = i % C;
  const int r  = (i / C) % R;
  const int bt = i / (R * C);
  out[(size_t)bt * R * C + (size_t)c * R + r] = (bf16)in[i];
}

// ---------------------------------------------------------------------------
// RoPE + [b,s,h*64+d] -> [b,h,s,d] bf16 repack (q and k)
// ---------------------------------------------------------------------------
__global__ void k_rope(const float* __restrict__ in, bf16* __restrict__ out, int n)
{
  const int idx = blockIdx.x * blockDim.x + threadIdx.x;
  if (idx >= n) return;
  const int col = idx % DMODEL;
  const int s   = (idx / DMODEL) % SEQ;
  const int b   = idx / (DMODEL * SEQ);
  const int h   = col >> 6, d = col & 63;
  const int j   = d & 31;
  const float inv = __expf(-(float)j * (9.210340371976184f / 32.0f)); // 10000^(-j/32)
  const float ang = (float)s * inv;
  float sn, cs;
  __sincosf(ang, &sn, &cs);
  const float xv  = in[idx];
  const float rot = (d < 32) ? -in[idx + 32] : in[idx - 32];
  out[(((size_t)b * NHEAD + h) * SEQ + s) * HDIM + d] = (bf16)(xv * cs + rot * sn);
}

// V repack: [b,s,h*64+d] f32 -> [b,h,d,s] bf16 (so P@V B-fragments are contiguous)
__global__ void k_vT(const float* __restrict__ in, bf16* __restrict__ out, int n)
{
  const int idx = blockIdx.x * blockDim.x + threadIdx.x;
  if (idx >= n) return;
  const int col = idx % DMODEL;
  const int s   = (idx / DMODEL) % SEQ;
  const int b   = idx / (DMODEL * SEQ);
  const int h   = col >> 6, d = col & 63;
  out[(((size_t)b * NHEAD + h) * HDIM + d) * SEQ + s] = (bf16)in[idx];
}

// ---------------------------------------------------------------------------
// MoE gating: logits = xn @ sel_w, then sigmoid + top-2 mask
// ---------------------------------------------------------------------------
__global__ void k_gate(const bf16* __restrict__ xn, const float* __restrict__ selw,
                       float* __restrict__ gl, int n)
{
  const int idx = blockIdx.x * blockDim.x + threadIdx.x;
  if (idx >= n) return;
  const int e = idx & 7, t = idx >> 3;
  const bf16* xr = xn + (size_t)t * DMODEL;
  float acc = 0.f;
  for (int i = 0; i < DMODEL; ++i) acc += (float)xr[i] * selw[(size_t)i * NEXP + e];
  gl[idx] = acc;
}

__global__ void k_topk(const float* __restrict__ gl, float* __restrict__ mg, int n)
{
  const int t = blockIdx.x * blockDim.x + threadIdx.x;
  if (t >= n) return;
  float gv[NEXP];
#pragma unroll
  for (int e = 0; e < NEXP; ++e) gv[e] = 1.f / (1.f + __expf(-gl[t * NEXP + e]));
  int i1 = 0; float v1 = gv[0];
#pragma unroll
  for (int e = 1; e < NEXP; ++e) if (gv[e] > v1) { v1 = gv[e]; i1 = e; }
  int i2 = -1; float v2 = -1.f;
#pragma unroll
  for (int e = 0; e < NEXP; ++e) if (e != i1 && gv[e] > v2) { v2 = gv[e]; i2 = e; }
#pragma unroll
  for (int e = 0; e < NEXP; ++e)
    mg[t * NEXP + e] = (e == i1 || e == i2) ? gv[e] : 0.f;
}

// ---------------------------------------------------------------------------
// Final: LN(lnf_t) on last token, LN(lnf_d), then @ out_w^T + out_b -> [2,8]
// block = 64 threads = 2 waves, wave b handles batch b
// ---------------------------------------------------------------------------
__global__ void k_head(const float* __restrict__ H,
                       const float* __restrict__ tg, const float* __restrict__ tb,
                       const float* __restrict__ dg, const float* __restrict__ db,
                       const float* __restrict__ ow, const float* __restrict__ ob,
                       float* __restrict__ out)
{
  __shared__ float ys[BATCH * DMODEL];
  const int wave = threadIdx.x >> 5, lane = threadIdx.x & 31;
  const int b = wave;
  const float* xr = H + ((size_t)b * SEQ + (SEQ - 1)) * DMODEL;
  float xs[32]; float s = 0.f;
#pragma unroll
  for (int j = 0; j < 32; ++j) { xs[j] = xr[lane + j * 32]; s += xs[j]; }
  for (int o = 16; o > 0; o >>= 1) s += __shfl_xor(s, o, 32);
  float mean = s * (1.f / DMODEL);
  float v = 0.f;
#pragma unroll
  for (int j = 0; j < 32; ++j) { float d = xs[j] - mean; v += d * d; }
  for (int o = 16; o > 0; o >>= 1) v += __shfl_xor(v, o, 32);
  float rs = rsqrtf(v * (1.f / DMODEL) + 1e-5f);
  float y1[32]; s = 0.f;
#pragma unroll
  for (int j = 0; j < 32; ++j) {
    const int i = lane + j * 32;
    y1[j] = (xs[j] - mean) * rs * tg[i] + tb[i];
    s += y1[j];
  }
  for (int o = 16; o > 0; o >>= 1) s += __shfl_xor(s, o, 32);
  mean = s * (1.f / DMODEL);
  v = 0.f;
#pragma unroll
  for (int j = 0; j < 32; ++j) { float d = y1[j] - mean; v += d * d; }
  for (int o = 16; o > 0; o >>= 1) v += __shfl_xor(v, o, 32);
  rs = rsqrtf(v * (1.f / DMODEL) + 1e-5f);
#pragma unroll
  for (int j = 0; j < 32; ++j) {
    const int i = lane + j * 32;
    ys[b * DMODEL + i] = (y1[j] - mean) * rs * dg[i] + db[i];
  }
  __syncthreads();
  const int t = threadIdx.x;
  if (t < BATCH * NOUTD) {
    const int b2 = t >> 3, o = t & 7;
    float acc = 0.f;
    for (int i = 0; i < DMODEL; ++i) acc += ys[b2 * DMODEL + i] * ow[(size_t)o * DMODEL + i];
    out[b2 * NOUTD + o] = acc + ob[o];
  }
}

// ===========================================================================
extern "C" void kernel_launch(void* const* d_in, const int* in_sizes, int n_in,
                              void* d_out, int out_size, void* d_ws, size_t ws_size,
                              hipStream_t stream)
{
  const float* x      = (const float*)d_in[0];
  const float* ln1_g  = (const float*)d_in[1];
  const float* ln1_b  = (const float*)d_in[2];
  const float* qw     = (const float*)d_in[3];
  const float* qb     = (const float*)d_in[4];
  const float* kw     = (const float*)d_in[5];
  const float* kb     = (const float*)d_in[6];
  const float* vw     = (const float*)d_in[7];
  const float* vb     = (const float*)d_in[8];
  const float* ow     = (const float*)d_in[9];
  const float* ob     = (const float*)d_in[10];
  const float* ln2_g  = (const float*)d_in[11];
  const float* ln2_b  = (const float*)d_in[12];
  const float* sel_w  = (const float*)d_in[13];
  const float* w1     = (const float*)d_in[14];
  const float* w2     = (const float*)d_in[15];
  const float* lnf_tg = (const float*)d_in[16];
  const float* lnf_tb = (const float*)d_in[17];
  const float* lnf_dg = (const float*)d_in[18];
  const float* lnf_db = (const float*)d_in[19];
  const float* in_w   = (const float*)d_in[20];
  const float* in_b   = (const float*)d_in[21];
  const float* out_w  = (const float*)d_in[22];
  const float* out_b  = (const float*)d_in[23];

  char* wsb = (char*)d_ws;
  size_t off = 0;
  auto carve = [&](size_t bytes) -> char* {
    char* p = wsb + off;
    off = (off + bytes + 255) & ~(size_t)255;
    return p;
  };
  const size_t WDD  = (size_t)NLAYER * DMODEL * DMODEL;           // per qkv/ow
  const size_t WEH  = (size_t)NLAYER * NEXP * DMODEL * EHID;      // w1/w2

  bf16*  WqT = (bf16*)carve(WDD * 2);
  bf16*  WkT = (bf16*)carve(WDD * 2);
  bf16*  WvT = (bf16*)carve(WDD * 2);
  bf16*  Wob = (bf16*)carve(WDD * 2);
  bf16*  W1T = (bf16*)carve(WEH * 2);
  bf16*  W2T = (bf16*)carve(WEH * 2);
  bf16*  Win = (bf16*)carve((size_t)DMODEL * DIN * 2);
  bf16*  Xbf = (bf16*)carve((size_t)NTOK * DIN * 2);
  float* Hbu = (float*)carve((size_t)NTOK * DMODEL * 4);
  bf16*  XN  = (bf16*) carve((size_t)NTOK * DMODEL * 2);
  float* TMP = (float*)carve((size_t)NTOK * DMODEL * 4);
  bf16*  Qbf = (bf16*) carve((size_t)NTOK * DMODEL * 2);
  bf16*  Kbf = (bf16*) carve((size_t)NTOK * DMODEL * 2);
  bf16*  VTb = (bf16*) carve((size_t)NTOK * DMODEL * 2);
  bf16*  Obf = (bf16*) carve((size_t)NTOK * DMODEL * 2);
  float* GL  = (float*)carve((size_t)NTOK * NEXP * 4);
  float* MG  = (float*)carve((size_t)NTOK * NEXP * 4);
  bf16*  MID = (bf16*) carve((size_t)NTOK * EHID * 2);

  const int TB = 256;
  auto blks = [](size_t n, int t) { return (unsigned)((n + t - 1) / t); };

  // ---- weight conversion (f32 -> bf16, transposed where B must be Bt[N][K])
  k_cvt_tr<<<blks(WDD, TB), TB, 0, stream>>>(qw, WqT, (int)WDD, DMODEL, DMODEL);
  k_cvt_tr<<<blks(WDD, TB), TB, 0, stream>>>(kw, WkT, (int)WDD, DMODEL, DMODEL);
  k_cvt_tr<<<blks(WDD, TB), TB, 0, stream>>>(vw, WvT, (int)WDD, DMODEL, DMODEL);
  k_cvt   <<<blks(WDD, TB), TB, 0, stream>>>(ow, Wob, (int)WDD); // y = o @ ow^T -> Bt = ow
  k_cvt_tr<<<blks(WEH, TB), TB, 0, stream>>>(w1, W1T, (int)WEH, DMODEL, EHID);
  k_cvt_tr<<<blks(WEH, TB), TB, 0, stream>>>(w2, W2T, (int)WEH, EHID, DMODEL);
  k_cvt   <<<blks((size_t)DMODEL * DIN, TB), TB, 0, stream>>>(in_w, Win, DMODEL * DIN);
  k_cvt   <<<blks((size_t)NTOK * DIN, TB), TB, 0, stream>>>(x, Xbf, NTOK * DIN);

  // ---- input projection: h = x @ in_w^T + in_b
  {
    dim3 g(DMODEL / 128, NTOK / 64);
    k_gemm<<<g, TB, 0, stream>>>(Xbf, Win, NTOK, DMODEL, DIN,
                                 in_b, nullptr, nullptr, 0, 0, Hbu, nullptr);
  }

  const unsigned nElem = blks((size_t)NTOK * DMODEL, TB);
  const size_t ATTN_LDS = 32 * 64 * 2 + 32 * 1024 * 4 + 32 * 1024 * 2; // 196 KB

  for (int l = 0; l < NLAYER; ++l) {
    const size_t lD  = (size_t)l * DMODEL;
    const size_t lDD = (size_t)l * DMODEL * DMODEL;

    // LN1 -> xn (bf16)
    k_ln<<<NTOK / 8, TB, 0, stream>>>(Hbu, ln1_g + lD, ln1_b + lD, XN);

    dim3 gD(DMODEL / 128, NTOK / 64);
    // Q
    k_gemm<<<gD, TB, 0, stream>>>(XN, WqT + lDD, NTOK, DMODEL, DMODEL,
                                  qb + lD, nullptr, nullptr, 0, 0, TMP, nullptr);
    k_rope<<<nElem, TB, 0, stream>>>(TMP, Qbf, NTOK * DMODEL);
    // K
    k_gemm<<<gD, TB, 0, stream>>>(XN, WkT + lDD, NTOK, DMODEL, DMODEL,
                                  kb + lD, nullptr, nullptr, 0, 0, TMP, nullptr);
    k_rope<<<nElem, TB, 0, stream>>>(TMP, Kbf, NTOK * DMODEL);
    // V
    k_gemm<<<gD, TB, 0, stream>>>(XN, WvT + lDD, NTOK, DMODEL, DMODEL,
                                  vb + lD, nullptr, nullptr, 0, 0, TMP, nullptr);
    k_vT<<<nElem, TB, 0, stream>>>(TMP, VTb, NTOK * DMODEL);

    // attention
    k_attn<<<BATCH * NHEAD * (SEQ / 32), TB, ATTN_LDS, stream>>>(Qbf, Kbf, VTb, Obf);

    // output projection + residual: h = h + o @ ow^T + ob
    k_gemm<<<gD, TB, 0, stream>>>(Obf, Wob + lDD, NTOK, DMODEL, DMODEL,
                                  ob + lD, Hbu, nullptr, 0, 0, Hbu, nullptr);

    // LN2 -> xn
    k_ln<<<NTOK / 8, TB, 0, stream>>>(Hbu, ln2_g + lD, ln2_b + lD, XN);

    // gating
    k_gate<<<blks((size_t)NTOK * NEXP, TB), TB, 0, stream>>>(XN, sel_w + (size_t)l * DMODEL * NEXP,
                                                            GL, NTOK * NEXP);
    k_topk<<<blks(NTOK, TB), TB, 0, stream>>>(GL, MG, NTOK);

    // MoE experts: h += (relu(xn @ w1[e]) * m[:,e]) @ w2[e]
    for (int e = 0; e < NEXP; ++e) {
      const size_t we = ((size_t)l * NEXP + e) * DMODEL * EHID;
      dim3 gU(EHID / 128, NTOK / 64);
      k_gemm<<<gU, TB, 0, stream>>>(XN, W1T + we, NTOK, EHID, DMODEL,
                                    nullptr, nullptr, MG + e, NEXP, 1,
                                    nullptr, MID);
      k_gemm<<<gD, TB, 0, stream>>>(MID, W2T + we, NTOK, DMODEL, EHID,
                                    nullptr, Hbu, nullptr, 0, 0, Hbu, nullptr);
    }
  }

  // final: LN_t(last) -> LN_d -> @ out_w^T + out_b
  k_head<<<1, 64, 0, stream>>>(Hbu, lnf_tg, lnf_tb, lnf_dg, lnf_db,
                               out_w, out_b, (float*)d_out);
  (void)in_sizes; (void)n_in; (void)out_size; (void)ws_size;
}
